// Attention_64742337020012
// MI455X (gfx1250) — matrix operand order repaired and checked by reading, not compiled, not run on hardware
//
#include <hip/hip_runtime.h>

typedef _Float16 v16h __attribute__((ext_vector_type(16)));
typedef _Float16 v8h  __attribute__((ext_vector_type(8)));
typedef float    v8f  __attribute__((ext_vector_type(8)));

#define B_    4
#define N_    2048
#define E_    768
#define H_    12
#define D_    64
#define RDIM_ 32

// ---------------------------------------------------------------------------
// Cross-lane helpers (DPP16, stays in VALU pipe; co-issues with WMMA).
// ---------------------------------------------------------------------------
template <int CTRL>
static __device__ __forceinline__ float dpp_f(float x) {
  int xi = __builtin_bit_cast(int, x);
  int r  = __builtin_amdgcn_update_dpp(xi, xi, CTRL, 0xf, 0xf, true);
  return __builtin_bit_cast(float, r);
}
static __device__ __forceinline__ float dpp_max16(float v) {
  v = fmaxf(v, dpp_f<0xB1>(v));   // quad_perm [1,0,3,2]  (xor 1)
  v = fmaxf(v, dpp_f<0x4E>(v));   // quad_perm [2,3,0,1]  (xor 2)
  v = fmaxf(v, dpp_f<0x124>(v));  // row_ror:4
  v = fmaxf(v, dpp_f<0x128>(v));  // row_ror:8
  return v;
}
static __device__ __forceinline__ float dpp_sum16(float v) {
  v += dpp_f<0xB1>(v);
  v += dpp_f<0x4E>(v);
  v += dpp_f<0x124>(v);
  v += dpp_f<0x128>(v);
  return v;
}
static __device__ __forceinline__ float rsum32(float v) {
#pragma unroll
  for (int m = 1; m < 32; m <<= 1) v += __shfl_xor(v, m, 32);
  return v;
}

static __device__ __forceinline__ v16h cat8(v8h lo, v8h hi) {
  return __builtin_shufflevector(lo, hi, 0, 1, 2, 3, 4, 5, 6, 7,
                                 8, 9, 10, 11, 12, 13, 14, 15);
}
static __device__ __forceinline__ v16h load_a_frag(const _Float16* p, int hf) {
  // A layout (16x32 f16): lane holds K = 8*hf + i  and  16 + 8*hf + i
  v8h lo = *(const v8h*)(p + hf * 8);
  v8h hi = *(const v8h*)(p + 16 + hf * 8);
  return cat8(lo, hi);
}
static __device__ __forceinline__ v16h load_b_frag(const _Float16* p, int hf) {
  // B layout (32x16 f16): lane holds K = 8*hf + i  and  16 + 8*hf + i
  v8h lo = *(const v8h*)(p + hf * 8);
  v8h hi = *(const v8h*)(p + 16 + hf * 8);
  return cat8(lo, hi);
}

// Async global -> LDS (ASYNCcnt-tracked, ISA 10.x). 16 bytes per op.
// LDS dest address is the workgroup-relative byte offset (generic LDS
// pointers carry it in the low 32 bits per the flat-aperture mapping).
static __device__ __forceinline__ unsigned lds_off(const void* p) {
  return (unsigned)(unsigned long long)p;
}
static __device__ __forceinline__ void async_b128(unsigned dst_lds,
                                                  const _Float16* src) {
  asm volatile("global_load_async_to_lds_b128 %0, %1, off"
               :: "v"(dst_lds), "v"(src) : "memory");
}

// ---------------------------------------------------------------------------
// Prep: per-head RMSNorm + xPos RoPE on q/k, fold 1/sqrt(D) into q, f32->f16,
// relayout [B,N,H*D] -> [B,H,N,D]. One wave per (b,h,n) row; lane owns 2 elems.
// ---------------------------------------------------------------------------
__global__ __launch_bounds__(256) void prep_qk_kernel(
    const float* __restrict__ q, const float* __restrict__ k,
    const float* __restrict__ qs, const float* __restrict__ ks,
    _Float16* __restrict__ qh, _Float16* __restrict__ kh) {
  int wid  = blockIdx.x * (blockDim.x >> 5) + (threadIdx.x >> 5);
  int lane = threadIdx.x & 31;
  int n = wid % N_;
  int h = (wid / N_) % H_;
  int b = wid / (N_ * H_);

  const float* qsrc = q + ((size_t)(b * N_ + n)) * E_ + h * D_;
  const float* ksrc = k + ((size_t)(b * N_ + n)) * E_ + h * D_;
  int d0 = lane * 2, d1 = d0 + 1;

  float q0 = qsrc[d0], q1 = qsrc[d1];
  float k0 = ksrc[d0], k1 = ksrc[d1];

  float ssq = rsum32(q0 * q0 + q1 * q1);
  float ssk = rsum32(k0 * k0 + k1 * k1);
  float rq = rsqrtf(ssq * (1.0f / D_) + 1e-6f);
  float rk = rsqrtf(ssk * (1.0f / D_) + 1e-6f);
  q0 *= qs[d0] * rq; q1 *= qs[d1] * rq;
  k0 *= ks[d0] * rk; k1 *= ks[d1] * rk;

  float t = (float)n;
  if (lane < 16) {  // rotary half: dims [0, RDIM)
    float j2  = (float)(2 * lane);
    float inv = __powf(10000.0f, -(j2 / (float)RDIM_));
    float fr  = t * inv;
    float s, c;
    __sincosf(fr, &s, &c);
    float base = (j2 + 0.4f * RDIM_) / (1.4f * RDIM_);
    float pw   = (t - (float)(N_ / 2)) * (1.0f / 512.0f);
    float sc   = __powf(base, pw);
    float isc  = 1.0f / sc;
    float qa = (q0 * c - q1 * s) * sc,  qb = (q1 * c + q0 * s) * sc;
    float ka = (k0 * c - k1 * s) * isc, kb = (k1 * c + k0 * s) * isc;
    q0 = qa; q1 = qb; k0 = ka; k1 = kb;
  }
  q0 *= 0.125f; q1 *= 0.125f;  // 1/sqrt(64) folded into q

  size_t dst = (((size_t)(b * H_ + h)) * N_ + n) * D_;
  qh[dst + d0] = (_Float16)q0; qh[dst + d1] = (_Float16)q1;
  kh[dst + d0] = (_Float16)k0; kh[dst + d1] = (_Float16)k1;
}

// V: [B,N,H*D] f32 -> TRANSPOSED [B,H,D,N] f16 (P·V B-fragments contiguous).
__global__ __launch_bounds__(256) void prep_v_kernel(const float* __restrict__ v,
                                                     _Float16* __restrict__ vt) {
  size_t idx = (size_t)blockIdx.x * blockDim.x + threadIdx.x;
  if (idx >= (size_t)B_ * N_ * E_) return;
  int e = (int)(idx % E_);
  int n = (int)((idx / E_) % N_);
  int b = (int)(idx / ((size_t)E_ * N_));
  int h = e / D_, d = e % D_;
  vt[(((size_t)(b * H_ + h)) * D_ + d) * N_ + n] = (_Float16)v[idx];
}

__global__ __launch_bounds__(256) void prep_w_kernel(const float* __restrict__ w,
                                                     _Float16* __restrict__ wh) {
  size_t idx = (size_t)blockIdx.x * blockDim.x + threadIdx.x;
  if (idx < (size_t)E_ * E_) wh[idx] = (_Float16)w[idx];
}

// ---------------------------------------------------------------------------
// Flash attention, block-cooperative:
//   4 waves/block = 4 consecutive 16-query tiles of the SAME (b,h).
//   K/V chunks staged once per block into double-buffered LDS via
//   global_load_async_to_lds_b128 (ASYNCcnt), then consumed by all waves.
//   QK^T and P·V via v_wmma_f32_16x16x32_f16; online softmax via DPP16.
// ---------------------------------------------------------------------------
__global__ __launch_bounds__(128) void attn_kernel(
    const _Float16* __restrict__ qh, const _Float16* __restrict__ kh,
    const _Float16* __restrict__ vt, _Float16* __restrict__ yh) {
  __shared__ __attribute__((aligned(64))) _Float16 Kl[2][32][64];  // 8 KB
  __shared__ __attribute__((aligned(64))) _Float16 Vl[2][64][32];  // 8 KB (V^T)
  __shared__ __attribute__((aligned(64))) _Float16 Plds[4][16][32];// 4 KB

  int tid   = threadIdx.x;
  int wslot = tid >> 5;
  int lane  = tid & 31;
  int hf    = lane >> 4;       // half-group: 0 or 1
  int lm    = lane & 15;

  const int TILES = N_ / 16;         // 128
  const int G     = TILES / 4;       // 32 tile-groups per (b,h)
  int grp = blockIdx.x;
  int qt4 = grp % G;
  int h   = (grp / G) % H_;
  int b   = grp / (G * H_);
  int qt  = qt4 * 4 + wslot;         // this wave's query tile

  const _Float16* Q = qh + (((size_t)(b * H_ + h)) * N_) * D_;
  const _Float16* K = kh + (((size_t)(b * H_ + h)) * N_) * D_;
  const _Float16* V = vt + (((size_t)(b * H_ + h)) * D_) * N_;  // [D][N]

  int qrow = qt * 16 + lm;
  v16h a0 = load_a_frag(Q + (size_t)qrow * D_, hf);       // d = 0..31
  v16h a1 = load_a_frag(Q + (size_t)qrow * D_ + 32, hf);  // d = 32..63

  float mrow[8], lrow[8];
  v8f O[4];
#pragma unroll
  for (int r = 0; r < 8; ++r) { mrow[r] = -1e30f; lrow[r] = 0.0f; }
#pragma unroll
  for (int t = 0; t < 4; ++t) { v8f z = {}; O[t] = z; }

  const int nfull = qt >> 1;           // per-wave: index of its diagonal chunk
  const int nmax  = 2 * qt4 + 1;       // block-uniform last chunk index

  // cooperative stage of one 32-key chunk into LDS buffer `buf`
  auto stage = [&](int chunk, int buf) {
    int kc = chunk * 32;
    // K rows kc..kc+31 are contiguous: 4 KB flat copy, 32 B per thread
    {
      unsigned dst = lds_off(&Kl[buf][0][0]) + (unsigned)tid * 16;
      const _Float16* g = K + (size_t)kc * D_ + tid * 8;
      async_b128(dst, g);
      async_b128(dst + 2048, g + 1024);
    }
    // V^T rows d=0..63, 64 B each: thread copies 32 B of row d = tid/2
    {
      int d = tid >> 1, part = (tid & 1) * 16;   // halves within row
      unsigned dst = lds_off(&Vl[buf][0][0]) + (unsigned)(d * 32 + part) * 2;
      const _Float16* g = V + (size_t)d * N_ + kc + part;
      async_b128(dst, g);
      async_b128(dst + 16, g + 8);
    }
  };

  stage(0, 0);                          // prologue
  for (int i = 0; i <= nmax; ++i) {
    int buf = i & 1;
    bool more = (i < nmax);             // block-uniform
    if (more) {
      stage(i + 1, buf ^ 1);
      // L2 prefetch two chunks ahead to feed the async engine
      if (i + 2 <= nmax)
        __builtin_prefetch(K + (size_t)(i + 2) * 32 * D_ + tid * 16, 0, 3);
      asm volatile("s_wait_asynccnt 0x4" ::: "memory");  // chunk i landed
    } else {
      asm volatile("s_wait_asynccnt 0x0" ::: "memory");
    }
    __syncthreads();                    // chunk i visible to all waves

    if (i <= nfull) {                   // wave-uniform predicate: EXEC stays ~0
      const int kc = i * 32;
      const _Float16* Kb = &Kl[buf][0][0];
      const _Float16* Vb = &Vl[buf][0][0];

      // ---- S = Q K^T on two 16-key tiles (K from LDS) -----------------
      v8f S[2];
#pragma unroll
      for (int kt = 0; kt < 2; ++kt) {
        const _Float16* kp = Kb + (kt * 16 + lm) * D_;
        v16h b0 = load_b_frag(kp, hf);
        v16h b1 = load_b_frag(kp + 32, hf);
        v8f c = {};
        c = __builtin_amdgcn_wmma_f32_16x16x32_f16(false, a0, false, b0, (short)0, c, false, false);
        c = __builtin_amdgcn_wmma_f32_16x16x32_f16(false, a1, false, b1, (short)0, c, false, false);
        S[kt] = c;
      }

      if (i == nfull) {                 // this wave's single masked chunk
#pragma unroll
        for (int kt = 0; kt < 2; ++kt) {
          int key = kc + kt * 16 + lm;
#pragma unroll
          for (int r = 0; r < 8; ++r) {
            int qr = qt * 16 + r + hf * 8;   // C layout: row = r + 8*hf
            if (key > qr) S[kt][r] = -1e30f;
          }
        }
      }

      // ---- online softmax (DPP16 reductions) --------------------------
#pragma unroll
      for (int r = 0; r < 8; ++r) {
        float mx   = dpp_max16(fmaxf(S[0][r], S[1][r]));
        float mnew = fmaxf(mrow[r], mx);
        float alpha = __expf(mrow[r] - mnew);
        float p0 = __expf(S[0][r] - mnew);
        float p1 = __expf(S[1][r] - mnew);
        float rs = dpp_sum16(p0 + p1);
        lrow[r] = lrow[r] * alpha + rs;
        mrow[r] = mnew;
#pragma unroll
        for (int t = 0; t < 4; ++t) O[t][r] *= alpha;
        Plds[wslot][r + hf * 8][lm]      = (_Float16)p0;  // C-layout -> LDS
        Plds[wslot][r + hf * 8][16 + lm] = (_Float16)p1;
      }

      asm volatile("s_wait_dscnt 0x0" ::: "memory");      // LDS RAW fence

      v8h plo = *(const v8h*)&Plds[wslot][lm][hf * 8];    // P in A layout
      v8h phi = *(const v8h*)&Plds[wslot][lm][16 + hf * 8];
      v16h pa = cat8(plo, phi);

      // ---- O += P V on four 16-wide d tiles (V from LDS) --------------
#pragma unroll
      for (int t = 0; t < 4; ++t) {
        v16h vb = load_b_frag(Vb + (t * 16 + lm) * 32, hf);
        O[t] = __builtin_amdgcn_wmma_f32_16x16x32_f16(false, pa, false, vb, (short)0, O[t], false, false);
      }
    }
    __syncthreads();                    // buf reusable at iteration i+2
  }

  // normalize and store f16 into [B,N,H,D] for the projection GEMM
#pragma unroll
  for (int r = 0; r < 8; ++r) {
    float inv_l = 1.0f / lrow[r];
    int n = qt * 16 + r + hf * 8;
#pragma unroll
    for (int t = 0; t < 4; ++t)
      yh[(((size_t)(b * N_ + n)) * H_ + h) * D_ + t * 16 + lm] =
          (_Float16)(O[t][r] * inv_l);
  }
}

// ---------------------------------------------------------------------------
// Projection: out[M=8192, E] = y(f16) @ W^T(f16) + bias, WMMA GEMM,
// one wave per 16x16 output tile, K loop over E in steps of 32.
// ---------------------------------------------------------------------------
__global__ __launch_bounds__(128) void proj_kernel(
    const _Float16* __restrict__ yh, const _Float16* __restrict__ wh,
    const float* __restrict__ bias, float* __restrict__ out) {
  int wslot = threadIdx.x >> 5;
  int lane  = threadIdx.x & 31;
  int hf    = lane >> 4;
  int lm    = lane & 15;

  const int NT = E_ / 16;                 // 48 col tiles
  int gw = blockIdx.x * 4 + wslot;
  int nt = gw % NT;
  int mt = gw / NT;

  int row = mt * 16 + lm;                 // A: M = lane%16
  int col = nt * 16 + lm;                 // B: N = lane%16 -> W row (out = y W^T)
  const _Float16* yp = yh + (size_t)row * E_;
  const _Float16* wp = wh + (size_t)col * E_;

  v8f acc = {};
  for (int kc = 0; kc < E_; kc += 32) {
    v16h a  = load_a_frag(yp + kc, hf);
    v16h bf = load_b_frag(wp + kc, hf);
    acc = __builtin_amdgcn_wmma_f32_16x16x32_f16(false, a, false, bf, (short)0, acc, false, false);
  }
#pragma unroll
  for (int r = 0; r < 8; ++r) {
    int orow = mt * 16 + r + hf * 8;      // C layout
    out[(size_t)orow * E_ + nt * 16 + lm] = acc[r] + bias[nt * 16 + lm];
  }
}

// ---------------------------------------------------------------------------
extern "C" void kernel_launch(void* const* d_in, const int* in_sizes, int n_in,
                              void* d_out, int out_size, void* d_ws, size_t ws_size,
                              hipStream_t stream) {
  const float* q  = (const float*)d_in[0];
  const float* k  = (const float*)d_in[1];
  const float* v  = (const float*)d_in[2];
  const float* qs = (const float*)d_in[3];
  const float* ks = (const float*)d_in[4];
  const float* pw = (const float*)d_in[5];
  const float* pb = (const float*)d_in[6];
  float* out = (float*)d_out;

  const size_t szQ = (size_t)B_ * H_ * N_ * D_;   // == B*N*E elements
  _Float16* qh = (_Float16*)d_ws;
  _Float16* kh = qh + szQ;
  _Float16* vt = kh + szQ;
  _Float16* yh = vt + szQ;
  _Float16* wh = yh + szQ;                        // E*E halves

  prep_qk_kernel<<<(B_ * H_ * N_) / 8, 256, 0, stream>>>(q, k, qs, ks, qh, kh);
  prep_v_kernel<<<((size_t)B_ * N_ * E_ + 255) / 256, 256, 0, stream>>>(v, vt);
  prep_w_kernel<<<((size_t)E_ * E_ + 255) / 256, 256, 0, stream>>>(pw, wh);

  // attention: B*H*(N/64) blocks, each = 4 waves on 4 tiles of one (b,h)
  attn_kernel<<<B_ * H_ * (N_ / 64), 128, 0, stream>>>(qh, kh, vt, yh);

  // projection: (B*N/16)*(E/16) = 24576 waves, 4 waves/block
  proj_kernel<<<((B_ * N_ / 16) * (E_ / 16)) / 4, 128, 0, stream>>>(yh, wh, pb, out);
}
